// HGATRecommender_84859963835144
// MI455X (gfx1250) — matrix operand order repaired and checked by reading, not compiled, not run on hardware
//
#include <hip/hip_runtime.h>
#include <hip/hip_bf16.h>

// ---------------- types ----------------
typedef __attribute__((ext_vector_type(16))) __bf16  v16bf;
typedef __attribute__((ext_vector_type(8)))  float   v8f;
typedef __attribute__((ext_vector_type(4)))  float   v4f;
typedef __attribute__((ext_vector_type(4)))  unsigned int v4u;

// ---------------- helpers ----------------
__device__ __forceinline__ unsigned short f2bfbits(float f) {
    __bf16 b = (__bf16)f;                      // native v_cvt (RNE)
    return __builtin_bit_cast(unsigned short, b);
}

// A fragment (16x32 bf16, wave32): lane holds row M=lane&15.
// elements 0..7  = A[row][kc+koff+0..7], elements 8..15 = A[row][kc+koff+16..23]
// koff = (lane>=16) ? 8 : 0
__device__ __forceinline__ v16bf load_a_frag(const float* __restrict__ p, float s) {
    v4f x0 = *(const v4f*)(p);
    v4f x1 = *(const v4f*)(p + 4);
    v4f x2 = *(const v4f*)(p + 16);
    v4f x3 = *(const v4f*)(p + 20);
    v16bf v;
#pragma unroll
    for (int i = 0; i < 4; ++i) {
        v[i]      = (__bf16)(x0[i] * s);
        v[4 + i]  = (__bf16)(x1[i] * s);
        v[8 + i]  = (__bf16)(x2[i] * s);
        v[12 + i] = (__bf16)(x3[i] * s);
    }
    return v;
}

// B fragment (32x16 bf16, wave32): lane holds col N=lane&15,
// element e<8 = B[kc + base + e][col], e>=8 = B[kc + base + 16 + (e-8)][col], base = (lane>=16)?8:0.
// Fragments are read from LDS (weights staged by async DMA) -> ds_load_b128 x2.
__device__ __forceinline__ v16bf load_b_frag_lds(const unsigned short* p) {
    union { v4u q[2]; v16bf v; } r;
    r.q[0] = *(const v4u*)(p);
    r.q[1] = *(const v4u*)(p + 16);
    return r.v;
}

// ---------------- GEMM kernel ----------------
// out = f( A_scaled @ B^T + bias [+ A2 @ B2^T] )
// 16 rows x 128 cols per wave, K = 128, weights staged to LDS via async DMA.
template <bool DUAL, bool L2N, bool ACCUM>
__global__ __launch_bounds__(256) void hgat_gemm(
    const float* __restrict__ A, const float* __restrict__ rowscale,
    const unsigned short* __restrict__ B, const float* __restrict__ bias,
    const float* __restrict__ A2, const unsigned short* __restrict__ B2,
    float* __restrict__ out, int nrows)
{
    __shared__ unsigned short sB[(DUAL ? 2 : 1) * 16384];   // 32KB / 64KB per block

    // ---- stage weight matrices into LDS with async DMA (16B per lane per op) ----
    {
        unsigned ldsbase = (unsigned)(unsigned long long)(&sB[0]);
#pragma unroll
        for (int j = 0; j < 8; ++j) {
            unsigned idx = (j * 256u + threadIdx.x) * 8u;       // ushort index
            unsigned laddr = ldsbase + idx * 2u;
            const unsigned short* g = B + idx;
            asm volatile("global_load_async_to_lds_b128 %0, %1, off"
                         :: "v"(laddr), "v"(g) : "memory");
        }
        if (DUAL) {
#pragma unroll
            for (int j = 0; j < 8; ++j) {
                unsigned idx = (j * 256u + threadIdx.x) * 8u;
                unsigned laddr = ldsbase + 32768u + idx * 2u;
                const unsigned short* g = B2 + idx;
                asm volatile("global_load_async_to_lds_b128 %0, %1, off"
                             :: "v"(laddr), "v"(g) : "memory");
            }
        }
        asm volatile("s_wait_asynccnt 0" ::: "memory");
        __syncthreads();
    }

    int wave = (int)((blockIdx.x * blockDim.x + threadIdx.x) >> 5);
    int row0 = wave << 4;
    if (row0 >= nrows) return;                 // after barrier: safe
    int lane  = threadIdx.x & 31;
    int half  = lane >> 4;                     // 0 | 1
    int lr    = lane & 15;
    int koff  = half << 3;                     // A K-offset: 0 | 8
    int bbase = half << 3;                     // B K-offset: 0 | 8
    int arow  = row0 + lr; if (arow >= nrows) arow = nrows - 1;

    float rs = rowscale ? rowscale[arow] : 1.0f;
    const float* Arow  = A + (size_t)arow * 128;
    const float* A2row = DUAL ? (A2 + (size_t)arow * 128) : nullptr;

    v8f acc[8];
#pragma unroll
    for (int t = 0; t < 8; ++t)
#pragma unroll
        for (int i = 0; i < 8; ++i) acc[t][i] = 0.0f;

#pragma unroll
    for (int kk = 0; kk < 4; ++kk) {
        const int kc = kk * 32;
        v16bf af = load_a_frag(Arow + kc + koff, rs);
        v16bf a2f;
        if (DUAL) a2f = load_a_frag(A2row + kc + koff, 1.0f);
#pragma unroll
        for (int t = 0; t < 8; ++t) {
            const int fo = (t * 16 + lr) * 128 + kc + bbase;
            v16bf bf1 = load_b_frag_lds(sB + fo);
            acc[t] = __builtin_amdgcn_wmma_f32_16x16x32_bf16(
                false, af, false, bf1, (short)0, acc[t], false, false);
            if (DUAL) {
                v16bf bf2 = load_b_frag_lds(sB + 16384 + fo);
                acc[t] = __builtin_amdgcn_wmma_f32_16x16x32_bf16(
                    false, a2f, false, bf2, (short)0, acc[t], false, false);
            }
        }
    }

    // bias per output column j = t*16 + lr (same for every row in this lane)
#pragma unroll
    for (int t = 0; t < 8; ++t) {
        float bv = bias[t * 16 + lr];
#pragma unroll
        for (int r = 0; r < 8; ++r) acc[t][r] += bv;
    }

    float scale[8];
    if (L2N) {
        // row (r + 8*half) lives in the 16 lanes of this half -> xor tree over 16 lanes
#pragma unroll
        for (int r = 0; r < 8; ++r) {
            float s = 0.0f;
#pragma unroll
            for (int t = 0; t < 8; ++t) s += acc[t][r] * acc[t][r];
            s += __shfl_xor(s, 1); s += __shfl_xor(s, 2);
            s += __shfl_xor(s, 4); s += __shfl_xor(s, 8);
            scale[r] = 1.0f / fmaxf(sqrtf(s), 1e-12f);
        }
    } else {
#pragma unroll
        for (int r = 0; r < 8; ++r) scale[r] = 1.0f;
    }

    // C/D layout: lane holds col lr of rows r (lanes 0-15) / r+8 (lanes 16-31)
#pragma unroll
    for (int r = 0; r < 8; ++r) {
        int grow = row0 + r + (half << 3);
        if (grow < nrows) {
            float* orow = out + (size_t)grow * 128;
#pragma unroll
            for (int t = 0; t < 8; ++t) {
                float v = acc[t][r] * scale[r];
                int c = t * 16 + lr;
                if (ACCUM) orow[c] += v; else orow[c] = v;
            }
        }
    }
}

// one wave per edge: scatter-add a 128-float feature row (4 floats/lane)
__global__ __launch_bounds__(256) void hgat_scatter(
    const float* __restrict__ hsrc, const int* __restrict__ src,
    const int* __restrict__ dst, int E, float* __restrict__ agg)
{
    long long tid = (long long)blockIdx.x * blockDim.x + threadIdx.x;
    int e = (int)(tid >> 5);
    if (e >= E) return;
    int lane = (int)(tid & 31);
    int s = src[e], d = dst[e];
    v4f v = *(const v4f*)(hsrc + (size_t)s * 128 + lane * 4);
    float* dp = agg + (size_t)d * 128 + lane * 4;
    unsafeAtomicAdd(dp + 0, v[0]);
    unsafeAtomicAdd(dp + 1, v[1]);
    unsafeAtomicAdd(dp + 2, v[2]);
    unsafeAtomicAdd(dp + 3, v[3]);
}

__global__ __launch_bounds__(256) void hgat_deg(
    const int* __restrict__ dst, int E, float* __restrict__ deg)
{
    int e = blockIdx.x * blockDim.x + threadIdx.x;
    if (e < E) unsafeAtomicAdd(deg + dst[e], 1.0f);
}

__global__ __launch_bounds__(256) void hgat_invdeg(float* __restrict__ deg, int n)
{
    int i = blockIdx.x * blockDim.x + threadIdx.x;
    if (i < n) deg[i] = 1.0f / fmaxf(deg[i], 1.0f);
}

__global__ __launch_bounds__(256) void hgat_cvt_bf16(
    const float* __restrict__ src, unsigned short* __restrict__ dst, int n)
{
    int i = blockIdx.x * blockDim.x + threadIdx.x;
    if (i < n) dst[i] = f2bfbits(src[i]);
}

// one wave per row: h = relu(layernorm(acc*invnd + h) * g + b), in place on h
__global__ __launch_bounds__(256) void hgat_ln(
    const float* __restrict__ acc, float* __restrict__ h,
    const float* __restrict__ g, const float* __restrict__ b,
    float invnd, int n)
{
    int row = (int)((blockIdx.x * blockDim.x + threadIdx.x) >> 5);
    if (row >= n) return;
    int lane = threadIdx.x & 31;
    size_t base = (size_t)row * 128 + lane * 4;
    v4f a  = *(const v4f*)(acc + base);
    v4f hv = *(const v4f*)(h + base);
    v4f x  = a * invnd + hv;
    float s = x[0] + x[1] + x[2] + x[3];
    s += __shfl_xor(s, 1); s += __shfl_xor(s, 2); s += __shfl_xor(s, 4);
    s += __shfl_xor(s, 8); s += __shfl_xor(s, 16);
    float m = s * (1.0f / 128.0f);
    v4f d = x - m;
    float ss = d[0]*d[0] + d[1]*d[1] + d[2]*d[2] + d[3]*d[3];
    ss += __shfl_xor(ss, 1); ss += __shfl_xor(ss, 2); ss += __shfl_xor(ss, 4);
    ss += __shfl_xor(ss, 8); ss += __shfl_xor(ss, 16);
    float rinv = rsqrtf(ss * (1.0f / 128.0f) + 1e-5f);
    v4f gg = *(const v4f*)(g + lane * 4);
    v4f bb = *(const v4f*)(b + lane * 4);
    v4f y = d * rinv * gg + bb;
#pragma unroll
    for (int i = 0; i < 4; ++i) y[i] = fmaxf(y[i], 0.0f);
    *(v4f*)(h + base) = y;
}

// ---------------- host orchestration ----------------
static const int NN[7]    = {100000, 40000, 3000, 20000, 8000, 60000, 3000};
static const int NOFF[7]  = {0, 100000, 140000, 143000, 163000, 171000, 231000};
static const int TOT_N    = 234000;
static const int EST[14]  = {0,1,1,2,1,3,1,3,1,4,1,5,5,6};
static const int EDT[14]  = {1,0,2,1,3,1,3,1,4,1,5,1,6,5};
static const int EC[14]   = {1000000,1000000,500000,500000,200000,200000,200000,200000,
                             100000,100000,300000,300000,400000,400000};
static const int DEGOFF[14] = {0,40000,140000,143000,183000,203000,243000,263000,
                               303000,311000,351000,411000,451000,454000};
static const int DEGTOT   = 514000;
static const float INV_ND[7] = {1.0f, 1.0f/6.0f, 1.0f, 0.5f, 1.0f, 0.5f, 1.0f};

extern "C" void kernel_launch(void* const* d_in, const int* in_sizes, int n_in,
                              void* d_out, int out_size, void* d_ws, size_t ws_size,
                              hipStream_t stream)
{
    (void)in_sizes; (void)n_in; (void)out_size; (void)ws_size;

    const float* x[7];
    for (int t = 0; t < 7; ++t) x[t] = (const float*)d_in[t];
    const int* ei[14];
    for (int e = 0; e < 14; ++e) ei[e] = (const int*)d_in[7 + e];
    const float* Wp  = (const float*)d_in[21];
    const float* bp  = (const float*)d_in[22];
    const float* Wl  = (const float*)d_in[23];
    const float* bl  = (const float*)d_in[24];
    const float* Wr  = (const float*)d_in[25];
    const float* lng = (const float*)d_in[26];
    const float* lnb = (const float*)d_in[27];
    const float* Wu  = (const float*)d_in[28];
    const float* bu  = (const float*)d_in[29];
    const float* Wv  = (const float*)d_in[30];
    const float* bv  = (const float*)d_in[31];
    float* out = (float*)d_out;

    // workspace layout (float offsets)
    float* ws  = (float*)d_ws;
    float* h   = ws;                         // 234000*128
    float* acc = ws + 29952000;              // 234000*128
    float* agg = ws + 59904000;              // 100000*128 (max dst)
    float* deg = ws + 72704000;              // 514000
    unsigned short* wb  = (unsigned short*)(ws + 73218000);
    unsigned short* WPb = wb;                // 7  * 16384
    unsigned short* WLb = wb + 114688;       // 42 * 16384
    unsigned short* WRb = wb + 802816;       // 42 * 16384
    unsigned short* WUb = wb + 1490944;      // 16384
    unsigned short* WVb = wb + 1507328;      // 16384

    // 1. convert all weights to bf16 (L2-resident afterwards)
    hgat_cvt_bf16<<<(7*16384 + 255)/256, 256, 0, stream>>>(Wp, WPb, 7*16384);
    hgat_cvt_bf16<<<(42*16384 + 255)/256, 256, 0, stream>>>(Wl, WLb, 42*16384);
    hgat_cvt_bf16<<<(42*16384 + 255)/256, 256, 0, stream>>>(Wr, WRb, 42*16384);
    hgat_cvt_bf16<<<(16384 + 255)/256, 256, 0, stream>>>(Wu, WUb, 16384);
    hgat_cvt_bf16<<<(16384 + 255)/256, 256, 0, stream>>>(Wv, WVb, 16384);

    // 2. inverse in-degree per edge type (constant across layers)
    hipMemsetAsync(deg, 0, (size_t)DEGTOT * 4, stream);
    for (int e = 0; e < 14; ++e)
        hgat_deg<<<(EC[e] + 255)/256, 256, 0, stream>>>(ei[e] + EC[e], EC[e], deg + DEGOFF[e]);
    hgat_invdeg<<<(DEGTOT + 255)/256, 256, 0, stream>>>(deg, DEGTOT);

    // 3. input projection: h[t] = x[t] @ Wp[t]^T + bp[t]
    for (int t = 0; t < 7; ++t)
        hgat_gemm<false,false,false><<<(NN[t] + 127)/128, 256, 0, stream>>>(
            x[t], nullptr, WPb + (size_t)t * 16384, bp + (size_t)t * 128,
            nullptr, nullptr, h + (size_t)NOFF[t] * 128, NN[t]);

    // 4. layers
    for (int l = 0; l < 3; ++l) {
        hipMemsetAsync(acc, 0, (size_t)TOT_N * 128 * 4, stream);
        for (int e = 0; e < 14; ++e) {
            const int st = EST[e], dt = EDT[e], E = EC[e], Nd = NN[dt];
            hipMemsetAsync(agg, 0, (size_t)Nd * 128 * 4, stream);
            long long thr = (long long)E * 32;
            hgat_scatter<<<(unsigned)((thr + 255)/256), 256, 0, stream>>>(
                h + (size_t)NOFF[st] * 128, ei[e], ei[e] + E, E, agg);
            // acc[dt] += l2norm( (agg*invdeg) @ Wl^T + bl + h[dt] @ Wr^T )
            hgat_gemm<true,true,true><<<(Nd + 127)/128, 256, 0, stream>>>(
                agg, deg + DEGOFF[e],
                WLb + (size_t)(l*14 + e) * 16384, bl + (size_t)(l*14 + e) * 128,
                h + (size_t)NOFF[dt] * 128, WRb + (size_t)(l*14 + e) * 16384,
                acc + (size_t)NOFF[dt] * 128, Nd);
        }
        for (int t = 0; t < 7; ++t)
            hgat_ln<<<(NN[t] + 7)/8, 256, 0, stream>>>(
                acc + (size_t)NOFF[t] * 128, h + (size_t)NOFF[t] * 128,
                lng + (size_t)(l*7 + t) * 128, lnb + (size_t)(l*7 + t) * 128,
                INV_ND[t], NN[t]);
    }

    // 5. output heads: l2norm(h @ W^T + b)
    hgat_gemm<false,true,false><<<(100000 + 127)/128, 256, 0, stream>>>(
        h, nullptr, WUb, bu, nullptr, nullptr, out, 100000);
    hgat_gemm<false,true,false><<<(40000 + 127)/128, 256, 0, stream>>>(
        h + (size_t)NOFF[1] * 128, nullptr, WVb, bv, nullptr, nullptr,
        out + (size_t)100000 * 128, 40000);
}
